// LSTM_75514114998549
// MI455X (gfx1250) — matrix operand order repaired and checked by reading, not compiled, not run on hardware
//
#include <hip/hip_runtime.h>

// ---------------------------------------------------------------------------
// Bidirectional LSTM head for MI455X (gfx1250, wave32, WMMA bf16).
//
// Only out[-1] = concat(hs_f[S-1], hs_b_scan[0]) reaches the output, so:
//   * forward LSTM: full 512-step scan (sequential, batch-parallel: 4 WGs
//     of 16 batch rows each, no inter-WG sync)
//   * backward LSTM: exactly ONE cell step from zero state on x[S-1]
//     (W_hh_b is multiplied by h=0 and is mathematically unused)
//
// Per step per WG: gates[16,1024] = [h(256)|x(128)] @ [W_hh|W_ih]^T, as
// 64 N-tiles x 12 K-blocks of v_wmma_f32_16x16x32_bf16. Wave w owns the
// i/f/g/o tiles of hidden tile w, keeps c in VGPRs for all 512 steps.
// The i/f half of W_hh (32 N-tiles, 264 KB bank-padded) is resident in the
// WGP's 320 KB LDS for the whole scan; the g/o half + W_ih stream from L2.
// The resident/streamed split is compile-time (g < 2) so no pointer selects
// (and hence no flat_load merges) are generated.
// ---------------------------------------------------------------------------

typedef __attribute__((ext_vector_type(16))) __bf16 bf16x16;
typedef __attribute__((ext_vector_type(8)))  __bf16 bf16x8;
typedef __attribute__((ext_vector_type(8)))  float  f32x8;

constexpr int S_LEN  = 512;
constexpr int BATCH  = 64;
constexpr int EDIM   = 128;   // embedding
constexpr int HDIM   = 256;   // hidden
constexpr int G4H    = 1024;  // 4*H gate rows
constexpr int HPAD   = HDIM + 8;  // +16B row pad: conflict-free ds_load_b128
constexpr int EPAD   = EDIM + 8;
constexpr int RES_NT = 32;    // resident W_hh N-tiles == i,f gates (512 rows)

// gfx1250 TRANS op; one independent op after it per the trans hazard (ISA 7.4).
__device__ __forceinline__ float fast_tanh(float x)
{
    float r;
    asm volatile("v_tanh_f32 %0, %1\n\tv_nop" : "=v"(r) : "v"(x));
    return r;
}
__device__ __forceinline__ float fast_sigm(float x)
{
    return 0.5f * fast_tanh(0.5f * x) + 0.5f;
}
__device__ __forceinline__ float sigm(float x) { return 1.0f / (1.0f + __expf(-x)); }

// --- f32 -> bf16 weight conversion (once per launch) ------------------------
__global__ void k_cvt_bf16(const float* __restrict__ in,
                           __bf16* __restrict__ out, int n)
{
    int i = blockIdx.x * blockDim.x + threadIdx.x;
    if (i < n) out[i] = (__bf16)in[i];
}

// --- forward LSTM scan: 1 workgroup (16 waves) per 16 batch rows ------------
__global__ __launch_bounds__(512, 1)
void k_lstm_fwd(const int*   __restrict__ seq,   const float* __restrict__ emb,
                const __bf16* __restrict__ Whh,  const __bf16* __restrict__ Wih,
                const float* __restrict__ bias4h, float* __restrict__ hfinal)
{
    __shared__ __bf16 wRes[RES_NT * 16][HPAD];  // resident W_hh rows 0..511
    __shared__ __bf16 hbuf[2][16][HPAD];        // h state, bf16, ping-pong
    __shared__ __bf16 xbuf[16][EPAD];           // this step's embeddings

    const int tid  = threadIdx.x;
    const int wave = tid >> 5;             // 16 waves (wave32)
    const int lane = tid & 31;
    const int l16  = lane & 15;
    const int hiK  = lane >> 4;            // K-half select: k = 16*hiK + s in both A and B
    const int b0   = blockIdx.x * 16;      // batch rows handled by this WG

    // ---- one-time: pin W_hh rows 0..RES_NT*16-1 into LDS (bank-padded) ----
    for (int idx = tid; idx < RES_NT * 16 * (HDIM / 8); idx += 512) {
        const int r  = idx >> 5;           // row
        const int ck = idx & 31;           // 8-element chunk within row
        *(bf16x8*)&wRes[r][ck * 8] = *(const bf16x8*)(Whh + r * HDIM + ck * 8);
    }
    // zero initial hidden state (ping buffer 0)
    for (int i = tid; i < 16 * HPAD; i += 512)
        ((__bf16*)hbuf[0])[i] = (__bf16)0.0f;

    int   nt[4];
    float bgs[4];
#pragma unroll
    for (int g = 0; g < 4; ++g) {
        nt[g]  = g * 16 + wave;                 // N-tile of gate g, hidden tile `wave`
        bgs[g] = bias4h[nt[g] * 16 + l16];      // bias depends on column only
    }

    f32x8 c;                                    // persistent cell state tile
#pragma unroll
    for (int j = 0; j < 8; ++j) c[j] = 0.0f;

    int ping = 0;
    __syncthreads();

    for (int s = 0; s < S_LEN; ++s) {
        // Opaque zero: makes weight addresses loop-variant so LICM cannot
        // hoist 48 B-tiles out of the scan (which spilled them to scratch).
        int zoff = 0;
        asm volatile("" : "+s"(zoff));

        // ---- gather embeddings for this step into LDS (bf16) ----
        {
            const int m   = wave;               // batch row 0..15
            const int e0  = lane * 4;           // 32 lanes * float4 = 128 floats
            const int tok = seq[s * BATCH + b0 + m];
            float4 xv = make_float4(0.f, 0.f, 0.f, 0.f);
            if (tok != 0) xv = *(const float4*)(emb + tok * EDIM + e0);
            __bf16* xp = &xbuf[m][e0];
            xp[0] = (__bf16)xv.x; xp[1] = (__bf16)xv.y;
            xp[2] = (__bf16)xv.z; xp[3] = (__bf16)xv.w;
        }
        __syncthreads();                        // x + previous h visible

        f32x8 acc[4];
#pragma unroll
        for (int g = 0; g < 4; ++g)
#pragma unroll
            for (int j = 0; j < 8; ++j) acc[g][j] = 0.0f;

        // pull this step's streamed (g/o gate) W panels toward the WGP
        __builtin_prefetch(Whh + (nt[2] * 16 + l16) * HDIM, 0, 1);
        __builtin_prefetch(Whh + (nt[3] * 16 + l16) * HDIM, 0, 1);

#pragma unroll
        for (int kb = 0; kb < 12; ++kb) {
            // ---- A operand: 16x32 bf16 tile of [h|x], rows = batch ----
            // ISA layout: lane L -> m = L%16; K chunks split by L>=16.
            const __bf16* arow;
            int k0;
            if (kb < 8) { arow = &hbuf[ping][l16][0]; k0 = kb * 32; }
            else        { arow = &xbuf[l16][0];       k0 = (kb - 8) * 32; }
            bf16x8 alo = *(const bf16x8*)(arow + k0 + hiK * 16);
            bf16x8 ahi = *(const bf16x8*)(arow + k0 + hiK * 16 + 8);
            bf16x16 A = __builtin_shufflevector(alo, ahi,
                          0,1,2,3,4,5,6,7,8,9,10,11,12,13,14,15);

#pragma unroll
            for (int g = 0; g < 4; ++g) {
                // B operand: col n of B = W row n (contiguous in K).
                // g<2 (i,f): LDS-resident -> ds_load.  g>=2 (g,o) and the
                // W_ih blocks: global stream -> global_load.  The split is
                // compile-time after unrolling, so no pointer select / flat.
                bf16x16 Bm;
                if (kb < 8) {
                    if (g < 2) {
                        Bm = *(const bf16x16*)(&wRes[nt[g] * 16 + l16][0]
                                               + zoff + kb * 32 + hiK * 16);
                    } else {
                        Bm = *(const bf16x16*)(Whh + zoff
                                               + (nt[g] * 16 + l16) * HDIM
                                               + kb * 32 + hiK * 16);
                    }
                } else {
                    Bm = *(const bf16x16*)(Wih + zoff
                                           + (nt[g] * 16 + l16) * EDIM
                                           + (kb - 8) * 32 + hiK * 16);
                }
                acc[g] = __builtin_amdgcn_wmma_f32_16x16x32_bf16(
                             false, A, false, Bm, (short)0, acc[g], false, false);
            }
        }

        // ---- LSTM cell update for hidden tile `wave` (all in registers) ----
        const int col = wave * 16 + l16;
#pragma unroll
        for (int j = 0; j < 8; ++j) {
            const int row = j + hiK * 8;        // C/D layout: M = vgpr + 8*(lane>=16)
            float iv = fast_sigm(acc[0][j] + bgs[0]);
            float fv = fast_sigm(acc[1][j] + bgs[1]);
            float gv = fast_tanh(acc[2][j] + bgs[2]);
            float ov = fast_sigm(acc[3][j] + bgs[3]);
            float cv = fv * c[j] + iv * gv;
            c[j] = cv;
            float hv = ov * fast_tanh(cv);
            hbuf[ping ^ 1][row][col] = (__bf16)hv;
            if (s == S_LEN - 1)
                hfinal[(b0 + row) * HDIM + col] = hv;
        }
        ping ^= 1;
        __syncthreads();                        // h complete; LDS reusable
    }
}

// --- backward one-step cell + output head -----------------------------------
// h_b = sigmoid(o) * tanh(sigmoid(i) * tanh(g)),  gates = x[S-1] @ W_ih_b^T + b_b
// out[b] = sigmoid( h_f . W_out[:256] + h_b . W_out[256:] + b_out )
__global__ __launch_bounds__(256)
void k_lstm_final(const int* __restrict__ seq,  const float* __restrict__ emb,
                  const float* __restrict__ Wihb, const float* __restrict__ bb,
                  const float* __restrict__ Wout, const float* __restrict__ bout,
                  const float* __restrict__ hfinal, float* __restrict__ out)
{
    __shared__ float xs[EDIM];
    __shared__ float red[256];
    const int b = blockIdx.x, t = threadIdx.x;

    if (t < EDIM) {
        int tok = seq[(S_LEN - 1) * BATCH + b];
        xs[t] = (tok != 0) ? emb[tok * EDIM + t] : 0.f;
    }
    __syncthreads();

    // f-gate multiplies c0 = 0 -> dead; compute i, g, o only.
    float di = bb[t], dg = bb[512 + t], dz = bb[768 + t];
    const float* wi = Wihb +  t        * EDIM;
    const float* wg = Wihb + (512 + t) * EDIM;
    const float* wo = Wihb + (768 + t) * EDIM;
    for (int e = 0; e < EDIM; ++e) {
        float x = xs[e];
        di += wi[e] * x; dg += wg[e] * x; dz += wo[e] * x;
    }
    float cv = sigm(di) * tanhf(dg);
    float hb = sigm(dz) * tanhf(cv);

    red[t] = hfinal[b * HDIM + t] * Wout[t] + hb * Wout[HDIM + t];
    __syncthreads();
    for (int off = 128; off > 0; off >>= 1) {
        if (t < off) red[t] += red[t + off];
        __syncthreads();
    }
    if (t == 0) out[b] = sigm(red[0] + bout[0]);
}

// ---------------------------------------------------------------------------
extern "C" void kernel_launch(void* const* d_in, const int* in_sizes, int n_in,
                              void* d_out, int out_size, void* d_ws, size_t ws_size,
                              hipStream_t stream)
{
    (void)in_sizes; (void)n_in; (void)out_size; (void)ws_size;
    const int*   seq  = (const int*)  d_in[0];
    const float* emb  = (const float*)d_in[1];
    const float* WihF = (const float*)d_in[2];
    const float* WhhF = (const float*)d_in[3];
    const float* bF   = (const float*)d_in[4];
    const float* WihB = (const float*)d_in[5];
    // d_in[6] = W_hh_b: unused (backward LSTM contributes only step 0, h=0)
    const float* bB   = (const float*)d_in[7];
    const float* Wo   = (const float*)d_in[8];
    const float* bO   = (const float*)d_in[9];
    float* out = (float*)d_out;

    char* ws = (char*)d_ws;
    __bf16* wsWhh = (__bf16*)(ws);                              // 512 KiB
    __bf16* wsWih = (__bf16*)(ws + (1 << 19));                  // 256 KiB
    float*  hfin  = (float*) (ws + (1 << 19) + (1 << 18));      // 64 KiB

    k_cvt_bf16<<<(G4H * HDIM + 255) / 256, 256, 0, stream>>>(WhhF, wsWhh, G4H * HDIM);
    k_cvt_bf16<<<(G4H * EDIM + 255) / 256, 256, 0, stream>>>(WihF, wsWih, G4H * EDIM);
    k_lstm_fwd<<<BATCH / 16, 512, 0, stream>>>(seq, emb, wsWhh, wsWih, bF, hfin);
    k_lstm_final<<<BATCH, 256, 0, stream>>>(seq, emb, WihB, bB, Wo, bO, hfin, out);
}
